// Attention_20985210208743
// MI455X (gfx1250) — matrix operand order repaired and checked by reading, not compiled, not run on hardware
//
#include <hip/hip_runtime.h>
#include <hip/hip_bf16.h>

// MI455X (gfx1250) single-head causal attention, B=4, S=2048, D=1024.
// All GEMMs via v_wmma_f32_16x16x32_bf16 (bf16 in, f32 accumulate).
// Intermediates (Q,K,V^T,attn-out in bf16 = 64MB) are L2-resident (192MB L2).

typedef __attribute__((ext_vector_type(16))) __bf16 bf16x16;
typedef __attribute__((ext_vector_type(8)))  __bf16 bf16x8;
typedef __attribute__((ext_vector_type(8)))  float  f32x8;
typedef __attribute__((ext_vector_type(4)))  float  f32x4;

constexpr int Bn = 4, Sn = 2048, Dn = 1024;
constexpr int Mtot = Bn * Sn;  // 8192

// ---- WMMA fragment helpers (layouts per CDNA5 ISA 7.12.2) ----------------

// A 16x32 bf16: lane<16 -> row=lane, K = {0..7, 16..23}; lane>=16 -> row=lane-16,
// K = {8..15, 24..31}.
__device__ inline bf16x16 load_a_bf16(const __bf16* base, int ld, int row0, int k0, int lane) {
  const int r  = row0 + (lane & 15);
  const int kk = k0 + ((lane >> 4) << 3);
  const __bf16* p = base + (size_t)r * ld + kk;
  bf16x8 lo = *(const bf16x8*)p;
  bf16x8 hi = *(const bf16x8*)(p + 16);
  bf16x16 a;
#pragma unroll
  for (int i = 0; i < 8; ++i) { a[i] = lo[i]; a[i + 8] = hi[i]; }
  return a;
}

__device__ inline bf16x16 load_a_f32(const float* base, int ld, int row0, int k0, int lane) {
  const int r  = row0 + (lane & 15);
  const int kk = k0 + ((lane >> 4) << 3);
  const float* p = base + (size_t)r * ld + kk;
  f32x4 f0 = *(const f32x4*)p;
  f32x4 f1 = *(const f32x4*)(p + 4);
  f32x4 f2 = *(const f32x4*)(p + 16);
  f32x4 f3 = *(const f32x4*)(p + 20);
  bf16x16 a;
#pragma unroll
  for (int i = 0; i < 4; ++i) {
    a[i]      = (__bf16)f0[i];
    a[i + 4]  = (__bf16)f1[i];
    a[i + 8]  = (__bf16)f2[i];
    a[i + 12] = (__bf16)f3[i];
  }
  return a;
}

// B 32x16 bf16 (source laid out N-major: Bsrc[n][k] contiguous in k):
// lane<16 -> col=lane, K = {0..7, 16..23}; lane>=16 -> col=lane-16, K = {8..15, 24..31}.
__device__ inline bf16x16 load_b_bf16(const __bf16* base, int ld, int col0, int k0, int lane) {
  const int c  = col0 + (lane & 15);
  const int kk = k0 + ((lane >> 4) << 3);
  const __bf16* p = base + (size_t)c * ld + kk;
  bf16x8 lo = *(const bf16x8*)p;
  bf16x8 hi = *(const bf16x8*)(p + 16);
  bf16x16 b;
#pragma unroll
  for (int i = 0; i < 8; ++i) { b[i] = lo[i]; b[i + 8] = hi[i]; }
  return b;
}

__device__ inline f32x8 wmma_bf16(bf16x16 a, bf16x16 b, f32x8 c) {
  return __builtin_amdgcn_wmma_f32_16x16x32_bf16(false, a, false, b, (short)0, c, false, false);
}

// ---- weight f32 -> bf16 conversion ---------------------------------------
__global__ __launch_bounds__(256) void cvt_f32_bf16_kernel(const float* __restrict__ src,
                                                           __bf16* __restrict__ dst, int n) {
  int i = blockIdx.x * blockDim.x + threadIdx.x;
  const int stride = gridDim.x * blockDim.x;
  for (; i < n; i += stride) dst[i] = (__bf16)src[i];
}

// ---- projection GEMM: out[m,n] = sum_k A[m,k] * W[n,k] + bias[n] ---------
// 8 waves/block; wave w does a 32x64 tile (M = blk.y*256 + w*32, N = blk.x*64).
// Per K-step: 2 A-frag + 4 B-frag loads feed 8 WMMAs (1.5 b128 loads / WMMA).
template <bool A_F32, bool OUT_F32, bool TRANS_OUT>
__global__ __launch_bounds__(256) void proj_gemm_kernel(const void* __restrict__ A_,
                                                        const __bf16* __restrict__ W,
                                                        const float* __restrict__ bias,
                                                        void* __restrict__ out_) {
  const int lane = threadIdx.x & 31;
  const int wave = threadIdx.x >> 5;
  const int n0 = blockIdx.x * 64;
  const int m0 = blockIdx.y * 256 + wave * 32;

  f32x8 acc[2][4] = {};
  for (int kt = 0; kt < Dn; kt += 32) {
    bf16x16 a[2];
#pragma unroll
    for (int i = 0; i < 2; ++i) {
      if constexpr (A_F32) a[i] = load_a_f32((const float*)A_, Dn, m0 + 16 * i, kt, lane);
      else                 a[i] = load_a_bf16((const __bf16*)A_, Dn, m0 + 16 * i, kt, lane);
    }
#pragma unroll
    for (int j = 0; j < 4; ++j) {
      bf16x16 b = load_b_bf16(W, Dn, n0 + j * 16, kt, lane);
      acc[0][j] = wmma_bf16(a[0], b, acc[0][j]);
      acc[1][j] = wmma_bf16(a[1], b, acc[1][j]);
    }
  }

  const int col = lane & 15, half = lane >> 4;
#pragma unroll
  for (int i = 0; i < 2; ++i) {
    const int mbase = m0 + 16 * i + half * 8;  // 8 consecutive rows per lane
#pragma unroll
    for (int j = 0; j < 4; ++j) {
      const int n = n0 + j * 16 + col;
      const float bv = bias[n];
      if constexpr (OUT_F32) {
        float* o = (float*)out_ + (size_t)mbase * Dn + n;
#pragma unroll
        for (int r = 0; r < 8; ++r) o[(size_t)r * Dn] = acc[i][j][r] + bv;
      } else if constexpr (TRANS_OUT) {
        const int bb = mbase >> 11, ss = mbase & (Sn - 1);  // V stored [b][d][s]
        bf16x8 v;
#pragma unroll
        for (int r = 0; r < 8; ++r) v[r] = (__bf16)(acc[i][j][r] + bv);
        *(bf16x8*)(((__bf16*)out_) + ((size_t)bb * Dn + n) * Sn + ss) = v;  // 16B store
      } else {
        __bf16* o = (__bf16*)out_ + (size_t)mbase * Dn + n;
#pragma unroll
        for (int r = 0; r < 8; ++r) o[(size_t)r * Dn] = (__bf16)(acc[i][j][r] + bv);
      }
    }
  }
}

// ---- attention: per (batch, 16-query tile); 4 waves; scores in dynamic LDS
__global__ __launch_bounds__(128) void attn_kernel(const __bf16* __restrict__ Q,
                                                   const __bf16* __restrict__ K,
                                                   const __bf16* __restrict__ Vt,
                                                   __bf16* __restrict__ O) {
  extern __shared__ float smem[];
  float* Sls  = smem;            // [16][ncols] raw scores -> exp values
  float* rinv = smem + 16 * Sn;  // [16] 1/rowsum

  const int b  = blockIdx.y;
  const int q0 = blockIdx.x * 16;
  const int ncols = ((q0 + 16) + 63) & ~63;  // causal bound, rounded to 64
  const int npair = ncols >> 6;

  const int tid = threadIdx.x;
  const int wave = tid >> 5;
  const int lane = tid & 31;

  const __bf16* Qb = Q  + (size_t)b * Sn * Dn;
  const __bf16* Kb = K  + (size_t)b * Sn * Dn;
  const __bf16* Vb = Vt + (size_t)b * Dn * Sn;   // [d][s]

  // Phase 1: S[16, ncols] = Q_tile @ K^T (no 1/sqrt(d) per reference), masked.
  // Each wave: 16x64 score tile -> Q A-fragment reused across 4 WMMAs.
  for (int kp = wave; kp < npair; kp += 4) {
    const int kv0 = kp * 64;
    f32x8 acc[4] = {};
    for (int d = 0; d < Dn; d += 32) {
      bf16x16 a = load_a_bf16(Qb, Dn, q0, d, lane);
#pragma unroll
      for (int j = 0; j < 4; ++j) {
        bf16x16 bm = load_b_bf16(Kb, Dn, kv0 + 16 * j, d, lane);  // col = key
        acc[j] = wmma_bf16(a, bm, acc[j]);
      }
    }
    const int col = lane & 15, half = lane >> 4;
#pragma unroll
    for (int j = 0; j < 4; ++j)
#pragma unroll
      for (int r = 0; r < 8; ++r) {
        const int row = r + half * 8;
        const int c = kv0 + 16 * j + col;
        Sls[row * ncols + c] = (c > q0 + row) ? -1e9f : acc[j][r];
      }
  }
  __syncthreads();

  // Phase 2: row softmax (8 consecutive lanes per row -> shfl_xor reductions)
  {
    const int row = tid >> 3;
    const int sub = tid & 7;
    float m = -3.0e38f;
    for (int c = sub; c < ncols; c += 8) m = fmaxf(m, Sls[row * ncols + c]);
    m = fmaxf(m, __shfl_xor(m, 1));
    m = fmaxf(m, __shfl_xor(m, 2));
    m = fmaxf(m, __shfl_xor(m, 4));
    float s = 0.f;
    for (int c = sub; c < ncols; c += 8) {
      const float e = __expf(Sls[row * ncols + c] - m);
      Sls[row * ncols + c] = e;
      s += e;
    }
    s += __shfl_xor(s, 1);
    s += __shfl_xor(s, 2);
    s += __shfl_xor(s, 4);
    if (sub == 0) rinv[row] = 1.0f / s;
  }
  __syncthreads();

  // Phase 3: O_tile = P @ V ; each LDS P A-fragment (f32->bf16) feeds 2 WMMAs.
  const int arow = lane & 15;
  const int ka = (lane >> 4) << 3;
  for (int np = wave; np < Dn / 32; np += 4) {
    const int n0 = np * 32;
    f32x8 acc0 = {}, acc1 = {};
    for (int kv = 0; kv < ncols; kv += 32) {
      const float* p = &Sls[arow * ncols + kv + ka];
      f32x4 f0 = *(const f32x4*)p;
      f32x4 f1 = *(const f32x4*)(p + 4);
      f32x4 f2 = *(const f32x4*)(p + 16);
      f32x4 f3 = *(const f32x4*)(p + 20);
      bf16x16 a;
#pragma unroll
      for (int i = 0; i < 4; ++i) {
        a[i]      = (__bf16)f0[i];
        a[i + 4]  = (__bf16)f1[i];
        a[i + 8]  = (__bf16)f2[i];
        a[i + 12] = (__bf16)f3[i];
      }
      bf16x16 b0 = load_b_bf16(Vb, Sn, n0, kv, lane);       // col = feature
      bf16x16 b1 = load_b_bf16(Vb, Sn, n0 + 16, kv, lane);
      acc0 = wmma_bf16(a, b0, acc0);
      acc1 = wmma_bf16(a, b1, acc1);
    }
    const int col = lane & 15, half = lane >> 4;
#pragma unroll
    for (int r = 0; r < 8; ++r) {
      const int row = r + half * 8;
      const size_t obase = ((size_t)b * Sn + q0 + row) * Dn + col;
      O[obase + n0]      = (__bf16)(acc0[r] * rinv[row]);
      O[obase + n0 + 16] = (__bf16)(acc1[r] * rinv[row]);
    }
  }
}

// ---- host ----------------------------------------------------------------
extern "C" void kernel_launch(void* const* d_in, const int* in_sizes, int n_in,
                              void* d_out, int out_size, void* d_ws, size_t ws_size,
                              hipStream_t stream) {
  (void)in_sizes; (void)n_in; (void)out_size; (void)ws_size;
  const float* x  = (const float*)d_in[0];
  const float* y  = (const float*)d_in[1];
  const float* z  = (const float*)d_in[2];
  // d_in[3] = mask (bool): causal, computed analytically in-kernel
  const float* Wq = (const float*)d_in[4];
  const float* bq = (const float*)d_in[5];
  const float* Wk = (const float*)d_in[6];
  const float* bk = (const float*)d_in[7];
  const float* Wv = (const float*)d_in[8];
  const float* bv = (const float*)d_in[9];
  const float* Wo = (const float*)d_in[10];
  const float* bo = (const float*)d_in[11];
  float* out = (float*)d_out;

  // workspace (bf16): Q, K, V^T, attn-out (16MB each), 4 weight copies (2MB each)
  __bf16* Qws  = (__bf16*)d_ws;
  __bf16* Kws  = Qws  + (size_t)Mtot * Dn;
  __bf16* Vtws = Kws  + (size_t)Mtot * Dn;
  __bf16* AOws = Vtws + (size_t)Mtot * Dn;
  __bf16* Wqb  = AOws + (size_t)Mtot * Dn;
  __bf16* Wkb  = Wqb + (size_t)Dn * Dn;
  __bf16* Wvb  = Wkb + (size_t)Dn * Dn;
  __bf16* Wob  = Wvb + (size_t)Dn * Dn;

  const int nW = Dn * Dn;
  cvt_f32_bf16_kernel<<<512, 256, 0, stream>>>(Wq, Wqb, nW);
  cvt_f32_bf16_kernel<<<512, 256, 0, stream>>>(Wk, Wkb, nW);
  cvt_f32_bf16_kernel<<<512, 256, 0, stream>>>(Wv, Wvb, nW);
  cvt_f32_bf16_kernel<<<512, 256, 0, stream>>>(Wo, Wob, nW);

  const dim3 gblk(256);
  const dim3 ggrid(Dn / 64, Mtot / 256);
  proj_gemm_kernel<true,  false, false><<<ggrid, gblk, 0, stream>>>(x, Wqb, bq, Qws);
  proj_gemm_kernel<true,  false, false><<<ggrid, gblk, 0, stream>>>(y, Wkb, bk, Kws);
  proj_gemm_kernel<true,  false, true ><<<ggrid, gblk, 0, stream>>>(z, Wvb, bv, Vtws);

  const size_t smem = (size_t)16 * Sn * sizeof(float) + 16 * sizeof(float);
  attn_kernel<<<dim3(Sn / 16, Bn), dim3(128), smem, stream>>>(Qws, Kws, Vtws, AOws);

  proj_gemm_kernel<false, true,  false><<<ggrid, gblk, 0, stream>>>(AOws, Wob, bo, out);
}
